// TransformerBlock_83245056131632
// MI455X (gfx1250) — matrix operand order repaired and checked by reading, not compiled, not run on hardware
//
#include <hip/hip_runtime.h>
#include <hip/hip_bf16.h>
#include <math.h>
#include <stdint.h>

typedef __attribute__((ext_vector_type(16))) __bf16 v16bf;
typedef __attribute__((ext_vector_type(8)))  __bf16 v8bf;
typedef __attribute__((ext_vector_type(8)))  float  v8f;

#define HW   65536
#define NTOT 262144

__device__ __forceinline__ unsigned short f2bf(float f) {
  unsigned u = __float_as_uint(f);
  unsigned r = (u + 0x7fffu + ((u >> 16) & 1u)) >> 16;  // round to nearest even
  return (unsigned short)r;
}

// ---------------------------------------------------------------------------
// fp32 -> bf16 weight conversion
// ---------------------------------------------------------------------------
__global__ void k_f2bf(const float* __restrict__ src, unsigned short* __restrict__ dst, int n) {
  int i = blockIdx.x * 256 + threadIdx.x;
  if (i < n) dst[i] = f2bf(src[i]);
}

// ---------------------------------------------------------------------------
// Channel LayerNorm (per pixel over C channels), optional elementwise v-gate,
// output bf16 in pixel-major [n][C] layout (K-contiguous for WMMA B-fragments)
// ---------------------------------------------------------------------------
__global__ void __launch_bounds__(256) k_ln(
    const float* __restrict__ in, const float* __restrict__ g, const float* __restrict__ bta,
    const float* __restrict__ vptr, long long vbs,
    unsigned short* __restrict__ outbf, int C) {
  int n = blockIdx.x * 256 + threadIdx.x;
  int b = n >> 16, pix = n & 65535;
  size_t base = ((size_t)(b * C) << 16) + pix;
  float s = 0.f, s2 = 0.f;
  for (int c = 0; c < C; ++c) {
    float x = in[base + ((size_t)c << 16)];
    s += x; s2 += x * x;
  }
  float inv = 1.0f / (float)C;
  float mu  = s * inv;
  float var = fmaxf(s2 * inv - mu * mu, 0.f);
  float rs  = rsqrtf(var + 1e-5f);
  size_t vbase = vptr ? ((size_t)b * vbs + pix) : 0;
  unsigned short* orow = outbf + (size_t)n * C;
  for (int c8 = 0; c8 < C; c8 += 8) {
    unsigned short tmp[8];
#pragma unroll
    for (int j = 0; j < 8; ++j) {
      int c = c8 + j;
      float x = in[base + ((size_t)c << 16)];
      float val = (x - mu) * rs * g[c] + bta[c];
      if (vptr) val *= vptr[vbase + ((size_t)c << 16)];
      tmp[j] = f2bf(val);
    }
    *(uint4*)(orow + c8) = *(uint4*)tmp;
  }
}

// ---------------------------------------------------------------------------
// WMMA GEMM: out[b,m,pix] = sum_k Wbf[m,k] * Xt[n,k]   (n = b*HW + pix)
// Wave computes 16(M) x 64(N); A fragment reused across 4 WMMAs per K-step.
// Optional residual add in epilogue (out may alias res for in-place update).
// ---------------------------------------------------------------------------
__global__ void __launch_bounds__(256) k_gemm(
    const unsigned short* __restrict__ Wbf,  // M x K bf16, row-major
    const unsigned short* __restrict__ Xt,   // NTOT x K bf16, row-major
    float* out,                              // (B, M, HW) fp32
    const float* res,                        // (B, M, HW) fp32 or null
    int M, int K) {
  int lane  = threadIdx.x & 31;
  int wave  = threadIdx.x >> 5;
  int n0    = blockIdx.x * 512 + wave * 64;
  int mtile = blockIdx.y;
  int half  = lane >> 4;
  int l16   = lane & 15;

  const unsigned short* arow = Wbf + (size_t)(mtile * 16 + l16) * K;
  v8f acc[4] = {v8f{}, v8f{}, v8f{}, v8f{}};

  for (int k0 = 0; k0 < K; k0 += 32) {
    // A fragment: 16x32 bf16. Lane half 0 holds K in {k0..k0+7, k0+16..k0+23}.
    v8bf alo = *(const v8bf*)(arow + k0 + half * 8);
    v8bf ahi = *(const v8bf*)(arow + k0 + 16 + half * 8);
    v16bf a;
#pragma unroll
    for (int t = 0; t < 8; ++t) { a[t] = alo[t]; a[t + 8] = ahi[t]; }
#pragma unroll
    for (int nt = 0; nt < 4; ++nt) {
      // B fragment: 32x16 bf16. Lane half 0 holds K in {k0..k0+7, k0+16..k0+23}.
      const unsigned short* brow = Xt + (size_t)(n0 + nt * 16 + l16) * K + k0 + half * 8;
      v8bf blo = *(const v8bf*)(brow);
      v8bf bhi = *(const v8bf*)(brow + 16);
      v16bf bb;
#pragma unroll
      for (int t = 0; t < 8; ++t) { bb[t] = blo[t]; bb[t + 8] = bhi[t]; }
      acc[nt] = __builtin_amdgcn_wmma_f32_16x16x32_bf16(
          false, a, false, bb, (short)0, acc[nt], false, false);
    }
  }
  // C/D layout: lane -> col N = lane&15, row M = v + 8*(lane>>4)
#pragma unroll
  for (int nt = 0; nt < 4; ++nt) {
    int nb  = n0 + nt * 16;
    int b   = nb >> 16;
    int pix = (nb & 65535) + l16;
#pragma unroll
    for (int v = 0; v < 8; ++v) {
      int mm = mtile * 16 + half * 8 + v;
      size_t o = ((size_t)(b * M + mm) << 16) + pix;
      float val = acc[nt][v];
      if (res) val += res[o];
      out[o] = val;
    }
  }
}

// ---------------------------------------------------------------------------
// Depthwise 3x3, SAME zero pad, NCHW (Ctot channels)
// ---------------------------------------------------------------------------
__global__ void __launch_bounds__(256) k_dw3(
    const float* __restrict__ in, const float* __restrict__ w,
    float* __restrict__ out, int Ctot) {
  long long idx = (long long)blockIdx.x * 256 + threadIdx.x;
  int plane = (int)(idx >> 16);
  int pix   = (int)(idx & 65535);
  int c = plane % Ctot;
  int y = pix >> 8, x = pix & 255;
  size_t base = (size_t)plane << 16;
  const float* wp = w + c * 9;
  float acc = 0.f;
#pragma unroll
  for (int dy = -1; dy <= 1; ++dy) {
    int yy = y + dy;
    if (yy < 0 || yy > 255) continue;
#pragma unroll
    for (int dx = -1; dx <= 1; ++dx) {
      int xx = x + dx;
      if (xx < 0 || xx > 255) continue;
      acc += in[base + yy * 256 + xx] * wp[(dy + 1) * 3 + (dx + 1)];
    }
  }
  out[base + pix] = acc;
}

// ---------------------------------------------------------------------------
// FSAS spectral attention: irfft2(rfft2(q)*rfft2(k)) == 8x8 circular conv.
// 4 patches per 256-thread block; q/k patches staged into LDS with CDNA5
// async-to-LDS loads (ASYNCcnt-tracked, VGPR-bypassing), then the 64-thread
// group computes the circular convolution out of LDS.
// ---------------------------------------------------------------------------
__global__ void __launch_bounds__(256) k_fsas_conv(
    const float* __restrict__ hid, float* __restrict__ attout) {
  __shared__ float Q[4][64];
  __shared__ float Kp[4][64];
  int g  = threadIdx.x >> 6;          // patch group within block
  int tl = threadIdx.x & 63;          // thread within group
  int bid   = blockIdx.x * 4 + g;     // b*128*1024 + c*1024 + patch
  int patch = bid & 1023;
  int c     = (bid >> 10) & 127;
  int b     = bid >> 17;
  int ph = patch >> 5, pw = patch & 31;
  int i = tl >> 3, j = tl & 7;

  size_t qbase = ((size_t)(b * 384 + c) << 16) + ph * 2048 + pw * 8;
  const float* qg = hid + qbase + i * 256 + j;
  const float* kg = qg + ((size_t)128 << 16);
  // Low 32 bits of a generic pointer to LDS are the wave-relative LDS offset.
  unsigned qlds = (unsigned)(uintptr_t)&Q[g][tl];
  unsigned klds = (unsigned)(uintptr_t)&Kp[g][tl];
  asm volatile("global_load_async_to_lds_b32 %0, %1, off"
               :: "v"(qlds), "v"(qg) : "memory");
  asm volatile("global_load_async_to_lds_b32 %0, %1, off"
               :: "v"(klds), "v"(kg) : "memory");
  asm volatile("s_wait_asynccnt 0x0" ::: "memory");
  __syncthreads();

  float acc = 0.f;
#pragma unroll
  for (int a = 0; a < 8; ++a)
#pragma unroll
    for (int bb = 0; bb < 8; ++bb)
      acc += Q[g][a * 8 + bb] * Kp[g][((i - a) & 7) * 8 + ((j - bb) & 7)];

  attout[((size_t)(b * 128 + c) << 16) + ph * 2048 + i * 256 + pw * 8 + j] = acc;
}

// ---------------------------------------------------------------------------
// DFFN spectral filter: y = irfft2(rfft2(x) * w), per 8x8 patch, in place.
// Real 8-point DFTs; imag of DC/Nyquist rows ignored (pocketfft C2R semantics).
// ---------------------------------------------------------------------------
__global__ void __launch_bounds__(256) k_dffn_fft(
    float* __restrict__ t, const float* __restrict__ fw) {
  const float C8[8] = {1.f, 0.70710678118654752f, 0.f, -0.70710678118654752f,
                       -1.f, -0.70710678118654752f, 0.f, 0.70710678118654752f};
  const float S8[8] = {0.f, 0.70710678118654752f, 1.f, 0.70710678118654752f,
                       0.f, -0.70710678118654752f, -1.f, -0.70710678118654752f};
  int gid = blockIdx.x * 256 + threadIdx.x;
  int patch = gid & 1023;
  int c = (gid >> 10) & 255;
  int b = gid >> 18;
  int ph = patch >> 5, pw = patch & 31;
  size_t base = ((size_t)(b * 256 + c) << 16) + ph * 2048 + pw * 8;

  float Zr[8][5], Zi[8][5];
#pragma unroll
  for (int u = 0; u < 8; ++u)
#pragma unroll
    for (int v = 0; v < 5; ++v) { Zr[u][v] = 0.f; Zi[u][v] = 0.f; }

  for (int i = 0; i < 8; ++i) {
    float row[8];
#pragma unroll
    for (int j = 0; j < 8; ++j) row[j] = t[base + i * 256 + j];
    float fr[5], fi[5];
#pragma unroll
    for (int v = 0; v < 5; ++v) {
      float ar = 0.f, ai = 0.f;
#pragma unroll
      for (int j = 0; j < 8; ++j) {
        ar += row[j] * C8[(v * j) & 7];
        ai -= row[j] * S8[(v * j) & 7];
      }
      fr[v] = ar; fi[v] = ai;
    }
#pragma unroll
    for (int u = 0; u < 8; ++u) {
      float cc = C8[(u * i) & 7], ss = S8[(u * i) & 7];
#pragma unroll
      for (int v = 0; v < 5; ++v) {
        Zr[u][v] += fr[v] * cc + fi[v] * ss;
        Zi[u][v] += fi[v] * cc - fr[v] * ss;
      }
    }
  }
  const float* wch = fw + c * 40;
#pragma unroll
  for (int u = 0; u < 8; ++u)
#pragma unroll
    for (int v = 0; v < 5; ++v) {
      float wv = wch[u * 5 + v];
      Zr[u][v] *= wv; Zi[u][v] *= wv;
    }
  for (int i = 0; i < 8; ++i) {
    float gr[5], gi[5];
#pragma unroll
    for (int v = 0; v < 5; ++v) {
      float ar = 0.f, ai = 0.f;
#pragma unroll
      for (int u = 0; u < 8; ++u) {
        float cc = C8[(u * i) & 7], ss = S8[(u * i) & 7];
        ar += Zr[u][v] * cc - Zi[u][v] * ss;
        ai += Zr[u][v] * ss + Zi[u][v] * cc;
      }
      gr[v] = ar * 0.125f; gi[v] = ai * 0.125f;
    }
#pragma unroll
    for (int j = 0; j < 8; ++j) {
      float y = gr[0] + ((j & 1) ? -gr[4] : gr[4]);
#pragma unroll
      for (int v = 1; v < 4; ++v)
        y += 2.f * (gr[v] * C8[(v * j) & 7] - gi[v] * S8[(v * j) & 7]);
      t[base + i * 256 + j] = y * 0.125f;
    }
  }
}

// ---------------------------------------------------------------------------
// Depthwise 3x3 on 256 ch + gelu(x1)*x2 gate -> bf16 pixel-major [n][128]
// ---------------------------------------------------------------------------
__device__ __forceinline__ float dw9(const float* __restrict__ p,
                                     const float* __restrict__ w, int y, int x) {
  float acc = 0.f;
#pragma unroll
  for (int dy = -1; dy <= 1; ++dy) {
    int yy = y + dy;
    if (yy < 0 || yy > 255) continue;
#pragma unroll
    for (int dx = -1; dx <= 1; ++dx) {
      int xx = x + dx;
      if (xx < 0 || xx > 255) continue;
      acc += p[yy * 256 + xx] * w[(dy + 1) * 3 + (dx + 1)];
    }
  }
  return acc;
}

__global__ void __launch_bounds__(256) k_dw_gate(
    const float* __restrict__ t, const float* __restrict__ w,
    unsigned short* __restrict__ gated) {
  int n = blockIdx.x * 256 + threadIdx.x;
  int b = n >> 16, pix = n & 65535;
  int y = pix >> 8, x = pix & 255;
  unsigned short* orow = gated + (size_t)n * 128;
  for (int cc = 0; cc < 16; ++cc) {
    unsigned short tmp[8];
#pragma unroll
    for (int j = 0; j < 8; ++j) {
      int c = cc * 8 + j;
      const float* p1 = t + ((size_t)(b * 256 + c) << 16);
      const float* p2 = p1 + ((size_t)128 << 16);
      float x1 = dw9(p1, w + c * 9, y, x);
      float x2 = dw9(p2, w + (c + 128) * 9, y, x);
      float ge = 0.5f * x1 * (1.f + erff(x1 * 0.70710678118654752f));
      tmp[j] = f2bf(ge * x2);
    }
    *(uint4*)(orow + cc * 8) = *(uint4*)tmp;
  }
}

// ---------------------------------------------------------------------------
extern "C" void kernel_launch(void* const* d_in, const int* in_sizes, int n_in,
                              void* d_out, int out_size, void* d_ws, size_t ws_size,
                              hipStream_t stream) {
  const float* x       = (const float*)d_in[0];
  const float* n1_w    = (const float*)d_in[1];
  const float* n1_b    = (const float*)d_in[2];
  const float* qkv_w   = (const float*)d_in[3];
  const float* att_dw  = (const float*)d_in[4];
  const float* att_nw  = (const float*)d_in[5];
  const float* att_nb  = (const float*)d_in[6];
  const float* att_po  = (const float*)d_in[7];
  const float* n2_w    = (const float*)d_in[8];
  const float* n2_b    = (const float*)d_in[9];
  const float* pi_w    = (const float*)d_in[10];
  const float* ffn_fft = (const float*)d_in[11];
  const float* ffn_dw  = (const float*)d_in[12];
  const float* po2_w   = (const float*)d_in[13];
  float* out = (float*)d_out;

  char* ws = (char*)d_ws;
  float* hid   = (float*)(ws);                           // 384*NTOT*4 = 402.7 MB
  float* hiddw = (float*)(ws + (size_t)402653184);       // 402.7 MB
  unsigned short* bfst = (unsigned short*)(ws + (size_t)805306368);  // 67.1 MB staging
  unsigned short* wbf  = (unsigned short*)(ws + (size_t)872415232);  // bf16 weights
  unsigned short* wqkv = wbf;
  unsigned short* wpo1 = wbf + 24576;
  unsigned short* wpi  = wbf + 32768;
  unsigned short* wpo2 = wbf + 49152;
  float* attout = hid;   // reuse: hid free after dwconv
  float* tbuf   = hid;   // reuse: attout free after LN-gate

  // weights -> bf16
  k_f2bf<<<96, 256, 0, stream>>>(qkv_w, wqkv, 24576);
  k_f2bf<<<32, 256, 0, stream>>>(att_po, wpo1, 8192);
  k_f2bf<<<64, 256, 0, stream>>>(pi_w, wpi, 16384);
  k_f2bf<<<32, 256, 0, stream>>>(po2_w, wpo2, 8192);

  // ---- FSAS branch ----
  k_ln<<<1024, 256, 0, stream>>>(x, n1_w, n1_b, nullptr, 0, bfst, 64);
  k_gemm<<<dim3(512, 24), 256, 0, stream>>>(wqkv, bfst, hid, nullptr, 384, 64);
  k_dw3<<<393216, 256, 0, stream>>>(hid, att_dw, hiddw, 384);
  k_fsas_conv<<<131072, 256, 0, stream>>>(hiddw, attout);
  k_ln<<<1024, 256, 0, stream>>>(attout, att_nw, att_nb,
                                 hiddw + ((size_t)256 << 16), (long long)384 << 16,
                                 bfst, 128);
  k_gemm<<<dim3(512, 4), 256, 0, stream>>>(wpo1, bfst, out, x, 64, 128);

  // ---- DFFN branch ----
  k_ln<<<1024, 256, 0, stream>>>(out, n2_w, n2_b, nullptr, 0, bfst, 64);
  k_gemm<<<dim3(512, 16), 256, 0, stream>>>(wpi, bfst, tbuf, nullptr, 256, 64);
  k_dffn_fft<<<4096, 256, 0, stream>>>(tbuf, ffn_fft);
  k_dw_gate<<<1024, 256, 0, stream>>>(tbuf, ffn_dw, bfst);
  k_gemm<<<dim3(512, 4), 256, 0, stream>>>(wpo2, bfst, out, out, 64, 128);
}
